// RNN_64055142252742
// MI455X (gfx1250) — matrix operand order repaired and checked by reading, not compiled, not run on hardware
//
#include <hip/hip_runtime.h>
#include <math.h>

// ---------------------------------------------------------------------------
// Vanilla RNN on MI455X (gfx1250):
//   xp = x @ Wx            (16384 x 1024 x 1024 GEMM, bf16 WMMA, f32 accum)
//   h_t = tanh(xp_t + h_{t-1} @ Wh)   (512 sequential 32x1024x1024 steps)
// Working set (~136MB + 36MB bf16 copies) is L2-resident (192MB L2).
// bf16 V_WMMA_F32_16X16X32_BF16 = 8x the per-instruction math of fp32 WMMA.
// x is converted to bf16 ONCE (prep) so the GEMM k-loop is conversion-free.
// ---------------------------------------------------------------------------

typedef __bf16 bf16;
typedef bf16  v16bf __attribute__((ext_vector_type(16)));
typedef bf16  v8bf  __attribute__((ext_vector_type(8)));
typedef bf16  v4bf  __attribute__((ext_vector_type(4)));
typedef float v8f   __attribute__((ext_vector_type(8)));
typedef int   v4i   __attribute__((ext_vector_type(4)));

#define DD   1024   // hidden / feature dim
#define BBB  32     // batch
#define LLL  512    // sequence length

__device__ __forceinline__ bf16 f2bf(float f) {
  // round-to-nearest-even f32 -> bf16
  unsigned u = __builtin_bit_cast(unsigned, f);
  unsigned r = (u + 0x7FFFu + ((u >> 16) & 1u)) >> 16;
  return __builtin_bit_cast(bf16, (unsigned short)r);
}

__device__ __forceinline__ void cvt4(v16bf& a, int base, float4 f) {
  a[base + 0] = f2bf(f.x); a[base + 1] = f2bf(f.y);
  a[base + 2] = f2bf(f.z); a[base + 3] = f2bf(f.w);
}

// exact identity: tanh(s) = 1 - 2/(exp(2s)+1); v_exp_f32 + v_rcp_f32
__device__ __forceinline__ float fast_tanh(float s) {
  float e = __expf(2.0f * s);
  return 1.0f - 2.0f * __builtin_amdgcn_rcpf(e + 1.0f);
}

// ---------------------------------------------------------------------------
// Kernel 0: prep. Transposed bf16 weights (B fragments become contiguous
// 32B-per-lane loads), bf16 h0 into ping buffer, zero barrier counter.
// ---------------------------------------------------------------------------
__global__ void rnn_prep(const float* __restrict__ Wx, const float* __restrict__ Wh,
                         const float* __restrict__ h0,
                         bf16* __restrict__ WxT, bf16* __restrict__ WhT,
                         bf16* __restrict__ hA, unsigned* __restrict__ cnt) {
  const int stride = gridDim.x * blockDim.x;
  const int gid = blockIdx.x * blockDim.x + threadIdx.x;
  for (int i = gid; i < DD * DD; i += stride) {
    const int k = i >> 10;
    const int n = i & (DD - 1);
    WxT[(n << 10) + k] = f2bf(Wx[i]);   // WxT[n][k] = Wx[k][n]
    WhT[(n << 10) + k] = f2bf(Wh[i]);
  }
  for (int i = gid; i < BBB * DD; i += stride) hA[i] = f2bf(h0[i]);
  if (gid == 0) *cnt = 0u;
}

// Kernel 0b: one-time x -> bf16 (removes all conversion from the GEMM k-loop)
__global__ void __launch_bounds__(256)
rnn_cvt_x(const float* __restrict__ x, bf16* __restrict__ xbf) {
  const int stride = gridDim.x * blockDim.x;
  const int gid = blockIdx.x * blockDim.x + threadIdx.x;
  const int n4 = (BBB * LLL * DD) / 4;
  const float4* src = reinterpret_cast<const float4*>(x);
  v4bf* dst = reinterpret_cast<v4bf*>(xbf);
  for (int i = gid; i < n4; i += stride) {
    float4 f = src[i];
    v4bf o;
    o[0] = f2bf(f.x); o[1] = f2bf(f.y); o[2] = f2bf(f.z); o[3] = f2bf(f.w);
    dst[i] = o;
  }
}

// ---------------------------------------------------------------------------
// Kernel 1 (fast path): xp = x_bf16 @ Wx, written into d_out.
// WG tile 64(M) x 256(N), 8 waves as 2x4; each wave = 32x64 (2x4 WMMA tiles).
// A layout (16x32 bf16): lane l: M = l%16, half=l/16,
//   elems[0..15] = K 16h..16h+15 (contiguous k, same order as B).
// B: 16 contiguous K bf16 per lane from pre-transposed WxT (L2 resident).
// ---------------------------------------------------------------------------
__global__ void __launch_bounds__(256)
rnn_xproj_gemm_bf(const bf16* __restrict__ xbf, const bf16* __restrict__ WxT,
                  float* __restrict__ out) {
  const int lane = threadIdx.x & 31;
  const int wave = threadIdx.x >> 5;
  const int lr   = lane & 15;
  const int half = lane >> 4;
  const int tn = blockIdx.x & 3;    // 4 N-tiles of 256
  const int tm = blockIdx.x >> 2;   // 256 M-tiles of 64
  const int m0 = tm * 64 + (wave >> 2) * 32;
  const int n0 = tn * 256 + (wave & 3) * 64;

  v8f acc[2][4] = {};

  for (int k0 = 0; k0 < DD; k0 += 32) {
    v16bf a[2], b[4];
#pragma unroll
    for (int mt = 0; mt < 2; ++mt) {
      const bf16* row = xbf + (size_t)(m0 + mt * 16 + lr) * DD + k0;
      v8bf lo = *reinterpret_cast<const v8bf*>(row + 16 * half);
      v8bf hi = *reinterpret_cast<const v8bf*>(row + 16 * half + 8);
      a[mt] = __builtin_shufflevector(lo, hi, 0, 1, 2, 3, 4, 5, 6, 7,
                                      8, 9, 10, 11, 12, 13, 14, 15);
    }
#pragma unroll
    for (int nt = 0; nt < 4; ++nt) {
      const bf16* col = WxT + (size_t)(n0 + nt * 16 + lr) * DD + k0 + 16 * half;
      b[nt] = *reinterpret_cast<const v16bf*>(col);
    }
#pragma unroll
    for (int mt = 0; mt < 2; ++mt)
#pragma unroll
      for (int nt = 0; nt < 4; ++nt)
        acc[mt][nt] = __builtin_amdgcn_wmma_f32_16x16x32_bf16(
            false, a[mt], false, b[nt], (short)0, acc[mt][nt], false, false);
  }

  // C/D layout: VGPR r -> M = r + 8*(lane>=16), N = lane%16
#pragma unroll
  for (int mt = 0; mt < 2; ++mt)
#pragma unroll
    for (int nt = 0; nt < 4; ++nt)
#pragma unroll
      for (int r = 0; r < 8; ++r) {
        const int row = m0 + mt * 16 + r + 8 * half;
        const int col = n0 + nt * 16 + lr;
        out[(size_t)row * DD + col] = acc[mt][nt][r];
      }
}

// Kernel 1 (fallback if ws too small for x_bf16): convert A in-register.
__global__ void __launch_bounds__(256)
rnn_xproj_gemm_f32(const float* __restrict__ x, const bf16* __restrict__ WxT,
                   float* __restrict__ out) {
  const int lane = threadIdx.x & 31;
  const int wave = threadIdx.x >> 5;
  const int lr   = lane & 15;
  const int half = lane >> 4;
  const int tn = blockIdx.x & 7;    // 8 N-tiles of 128
  const int tm = blockIdx.x >> 3;   // 256 M-tiles of 64
  const int m0 = tm * 64 + (wave >> 2) * 32;
  const int n0 = tn * 128 + (wave & 3) * 32;

  v8f acc[2][2] = {};

  for (int k0 = 0; k0 < DD; k0 += 32) {
    v16bf a[2], b[2];
#pragma unroll
    for (int mt = 0; mt < 2; ++mt) {
      const float* row = x + (size_t)(m0 + mt * 16 + lr) * DD + k0;
      const float4* p0 = reinterpret_cast<const float4*>(row + 16 * half);
      const float4* p1 = reinterpret_cast<const float4*>(row + 16 * half + 8);
      float4 f0 = p0[0], f1 = p0[1];
      float4 f2 = p1[0], f3 = p1[1];
      cvt4(a[mt], 0, f0);  cvt4(a[mt], 4, f1);
      cvt4(a[mt], 8, f2);  cvt4(a[mt], 12, f3);
    }
#pragma unroll
    for (int nt = 0; nt < 2; ++nt) {
      const bf16* col = WxT + (size_t)(n0 + nt * 16 + lr) * DD + k0 + 16 * half;
      b[nt] = *reinterpret_cast<const v16bf*>(col);
    }
#pragma unroll
    for (int mt = 0; mt < 2; ++mt)
#pragma unroll
      for (int nt = 0; nt < 2; ++nt)
        acc[mt][nt] = __builtin_amdgcn_wmma_f32_16x16x32_bf16(
            false, a[mt], false, b[nt], (short)0, acc[mt][nt], false, false);
  }
#pragma unroll
  for (int mt = 0; mt < 2; ++mt)
#pragma unroll
    for (int nt = 0; nt < 2; ++nt)
#pragma unroll
      for (int r = 0; r < 8; ++r) {
        const int row = m0 + mt * 16 + r + 8 * half;
        const int col = n0 + nt * 16 + lr;
        out[(size_t)row * DD + col] = acc[mt][nt][r];
      }
}

// ---------------------------------------------------------------------------
// Kernel 2: sequential recurrence. Persistent, 16 resident WGs, each owns 64
// output columns. Per step: stage full h_{t-1} (64KB bf16) into LDS — via the
// CDNA5 async global->LDS path (ASYNCcnt) when the builtin exists — then each
// of 8 waves does one 16x16 tile over K=1024 (32 bf16 WMMAs), adds xp
// (in-place in d_out), tanh, writes fp32 out + bf16 h_t, then crosses a
// device-wide atomic barrier (monotonic counter, reset by prep each launch).
// ---------------------------------------------------------------------------
__global__ void __launch_bounds__(256)
rnn_recurrence(const bf16* __restrict__ WhT, bf16* __restrict__ hA,
               bf16* __restrict__ hB, float* __restrict__ out,
               unsigned* __restrict__ cnt) {
  __shared__ bf16 sh[BBB * DD];   // 64 KB: full h_{t-1}

  const int lane = threadIdx.x & 31;
  const int lr   = lane & 15;
  const int half = lane >> 4;
  const int wave = threadIdx.x >> 5;
  const int m0   = (wave >> 2) * 16;                 // 0 or 16
  const int ncol = blockIdx.x * 64 + (wave & 3) * 16;
  const unsigned nWG = gridDim.x;                    // 16

  for (int t = 0; t < LLL; ++t) {
    const bf16* hp = (t & 1) ? hB : hA;
    bf16*       hn = (t & 1) ? hA : hB;

    __threadfence();  // acquire h_{t-1} written by other WGs
#if __has_builtin(__builtin_amdgcn_global_load_async_to_lds_b128)
    {
      // param0 is 'int4 __device__*' (AS1) per clang's diagnostic; LDS side AS3.
      typedef __attribute__((address_space(1))) v4i g_v4i;
      typedef __attribute__((address_space(3))) v4i l_v4i;
      g_v4i* gsrc = (g_v4i*)hp;
      l_v4i* ldst = (l_v4i*)sh;
      for (int i = threadIdx.x; i < (BBB * DD * 2) / 16; i += 256)
        __builtin_amdgcn_global_load_async_to_lds_b128(gsrc + i, ldst + i, 0, 0);
#if __has_builtin(__builtin_amdgcn_s_wait_asynccnt)
      __builtin_amdgcn_s_wait_asynccnt(0);
#else
      asm volatile("s_wait_asynccnt 0x0" ::: "memory");
#endif
    }
#else
    {
      const uint4* src = reinterpret_cast<const uint4*>(hp);
      uint4*       dst = reinterpret_cast<uint4*>(sh);
      for (int i = threadIdx.x; i < (BBB * DD * 2) / 16; i += 256)
        dst[i] = src[i];
    }
#endif
    __syncthreads();

    v8f acc = {};
    for (int k0 = 0; k0 < DD; k0 += 32) {
      const bf16* ar = sh + (m0 + lr) * DD + k0;
      v8bf lo = *reinterpret_cast<const v8bf*>(ar + 16 * half);
      v8bf hi = *reinterpret_cast<const v8bf*>(ar + 16 * half + 8);
      v16bf a = __builtin_shufflevector(lo, hi, 0, 1, 2, 3, 4, 5, 6, 7,
                                        8, 9, 10, 11, 12, 13, 14, 15);
      v16bf b = *reinterpret_cast<const v16bf*>(
          WhT + (size_t)(ncol + lr) * DD + k0 + 16 * half);
      acc = __builtin_amdgcn_wmma_f32_16x16x32_bf16(
          false, a, false, b, (short)0, acc, false, false);
    }

#pragma unroll
    for (int r = 0; r < 8; ++r) {
      const int m = m0 + r + 8 * half;    // batch index (M == B == 32)
      const int c = ncol + lr;
      const size_t idx = ((size_t)m * LLL + t) * DD + c;
      const float v = fast_tanh(acc[r] + out[idx]);  // out currently holds xp
      out[idx] = v;
      hn[m * DD + c] = f2bf(v);
    }

    // device-wide barrier across the 16 resident workgroups
    __syncthreads();
    if (threadIdx.x == 0) {
      __threadfence();
      atomicAdd(cnt, 1u);
      const unsigned goal = nWG * (unsigned)(t + 1);
      while (atomicAdd(cnt, 0u) < goal) __builtin_amdgcn_s_sleep(2);
    }
    __syncthreads();
  }
}

// ---------------------------------------------------------------------------
extern "C" void kernel_launch(void* const* d_in, const int* in_sizes, int n_in,
                              void* d_out, int out_size, void* d_ws, size_t ws_size,
                              hipStream_t stream) {
  const float* x  = (const float*)d_in[0];   // [32, 512, 1024]
  const float* h0 = (const float*)d_in[1];   // [32, 1024]
  const float* Wx = (const float*)d_in[2];   // [1024, 1024]
  const float* Wh = (const float*)d_in[3];   // [1024, 1024]
  float* out = (float*)d_out;                // [32, 512, 1024]

  char* w = (char*)d_ws;
  bf16* WxT = (bf16*)(w);                                  // 2 MB
  bf16* WhT = (bf16*)(w + (size_t)2 * 1024 * 1024);        // 2 MB
  bf16* hA  = (bf16*)(w + (size_t)4 * 1024 * 1024);        // 64 KB
  bf16* hB  = (bf16*)(w + (size_t)4 * 1024 * 1024 + 65536);
  unsigned* cnt = (unsigned*)(w + (size_t)4 * 1024 * 1024 + 2 * 65536);
  bf16* xbf = (bf16*)(w + (size_t)8 * 1024 * 1024);        // 32 MB

  const size_t need = (size_t)40 * 1024 * 1024;

  rnn_prep<<<dim3(1024), dim3(256), 0, stream>>>(Wx, Wh, h0, WxT, WhT, hA, cnt);
  if (ws_size >= need) {
    rnn_cvt_x<<<dim3(4096), dim3(256), 0, stream>>>(x, xbf);
    rnn_xproj_gemm_bf<<<dim3(1024), dim3(256), 0, stream>>>(xbf, WxT, out);
  } else {
    rnn_xproj_gemm_f32<<<dim3(2048), dim3(256), 0, stream>>>(x, WxT, out);
  }
  rnn_recurrence<<<dim3(16), dim3(256), 0, stream>>>(WhT, hA, hB, out, cnt);
}
